// Stage2LagrangianStateSpaceModel_86028194939148
// MI455X (gfx1250) — matrix operand order repaired and checked by reading, not compiled, not run on hardware
//
#include <hip/hip_runtime.h>
#include <math.h>

#define D 128
#define TSTEPS 256
#define NB 8

typedef _Float16 v16h __attribute__((ext_vector_type(16)));
typedef _Float16 v8h  __attribute__((ext_vector_type(8)));
typedef float    v8f  __attribute__((ext_vector_type(8)));

__device__ __forceinline__ float san(float x) {
    x = (x == x) ? x : 0.0f;
    return fminf(fmaxf(x, -1.0e6f), 1.0e6f);
}
__device__ __forceinline__ float clamp_h(float x) {   // keep finite in f16
    return fminf(fmaxf(x, -65504.0f), 65504.0f);
}
__device__ __forceinline__ _Float16 to_h(float x) { return (_Float16)clamp_h(x); }

// C = A * B where:
//   A  : f16 row-major (LDS), rows are the M dim
//   Bt : f16 row-major storage whose row n supplies logical B[k][n]
//        (i.e. pass B^T; for symmetric B pass B itself) -> contiguous loads
// Output: optional f16 row-major (outH), optional f32 (outF, accF accumulates),
// optional f32 C-initializer addC (e.g. +Q), optional san on result.
// All fragment traffic is 16-byte ds_load_b128s; no per-element converts.
__device__ __forceinline__ void mm128h(const _Float16* __restrict__ A,
                                       const _Float16* __restrict__ Bt,
                                       _Float16* __restrict__ outH,
                                       float* __restrict__ outF,
                                       const float* __restrict__ addC,
                                       bool accF, bool sanOut)
{
    const int tid  = threadIdx.x;
    const int lane = tid & 31;
    const int tm   = tid >> 5;        // wave id = output tile-row block
    const int hl   = lane >> 4;
    const int lm   = lane & 15;
    const int arow = tm * 16 + lm;

    #pragma unroll
    for (int tn = 0; tn < 8; ++tn) {
        const int brow = tn * 16 + lm;
        v8f acc;
        if (accF) {
            #pragma unroll
            for (int r = 0; r < 8; ++r)
                acc[r] = outF[(tm * 16 + r + 8 * hl) * D + tn * 16 + lm];
        } else if (addC) {
            #pragma unroll
            for (int r = 0; r < 8; ++r)
                acc[r] = addC[(tm * 16 + r + 8 * hl) * D + tn * 16 + lm];
        } else {
            #pragma unroll
            for (int r = 0; r < 8; ++r) acc[r] = 0.0f;
        }
        #pragma unroll
        for (int kb = 0; kb < 4; ++kb) {
            // A frag (ISA 16-bit A layout): k = {8*hl..8*hl+7} and {16+8*hl..}
            v8h a0 = *(const v8h*)(A + arow * D + kb * 32 + 8 * hl);
            v8h a1 = *(const v8h*)(A + arow * D + kb * 32 + 8 * hl + 16);
            // B frag (same layout as A): k = {8*hl..8*hl+7} and {16+8*hl..} in row `brow` of Bt
            v8h b0 = *(const v8h*)(Bt + brow * D + kb * 32 + 8 * hl);
            v8h b1 = *(const v8h*)(Bt + brow * D + kb * 32 + 8 * hl + 16);
            v16h af = __builtin_shufflevector(a0, a1, 0,1,2,3,4,5,6,7,8,9,10,11,12,13,14,15);
            v16h bf = __builtin_shufflevector(b0, b1, 0,1,2,3,4,5,6,7,8,9,10,11,12,13,14,15);
            acc = __builtin_amdgcn_wmma_f32_16x16x32_f16(false, af, false, bf,
                                                         (short)0, acc, false, false);
        }
        #pragma unroll
        for (int r = 0; r < 8; ++r) {
            const int row = tm * 16 + r + 8 * hl, col = tn * 16 + lm;
            float v = sanOut ? san(acc[r]) : acc[r];
            if (outF) outF[row * D + col] = v;
            if (outH) outH[row * D + col] = to_h(v);
        }
    }
}

// Q = san(q q^T / D)+1e-6 I, R likewise; also emit f16 row-major R mirror.
__global__ void kf_build_qr(const float* __restrict__ qf, const float* __restrict__ rf,
                            float* __restrict__ Qg, float* __restrict__ Rg,
                            _Float16* __restrict__ Rh)
{
    int gid = blockIdx.x * blockDim.x + threadIdx.x;   // 0..32767
    int which = gid >> 14;
    int idx = gid & 16383;
    int i = idx >> 7, j = idx & (D - 1);
    const float* f = which ? rf : qf;
    float s = 0.0f;
    for (int k = 0; k < D; ++k) s += f[i * D + k] * f[j * D + k];
    float v = san(s * (1.0f / D));
    if (i == j) v += 1.0e-6f;
    if (which) { Rg[idx] = v; Rh[idx] = to_h(v); }
    else       { Qg[idx] = v; }
}

__global__ void kf_filter(const float* __restrict__ obs,
                          const float* __restrict__ Fraw,
                          const float* __restrict__ im,
                          const float* __restrict__ ild,
                          const float* __restrict__ praw,
                          const float* __restrict__ graw,
                          const float* __restrict__ Qg,
                          const float* __restrict__ Rg,
                          const _Float16* __restrict__ Rh,
                          float* __restrict__ out)
{
    extern __shared__ float smem[];
    float* Pm = smem;                       // P state (f32, persistent)
    float* Lb = smem + 16384;               // S -> Cholesky L (f32)
    float* Kf = smem + 2 * 16384;           // K (f32, solve output/scratch)
    _Float16* H0 = (_Float16*)(smem + 3 * 16384);   // Fh -> IKh
    _Float16* H1 = H0 + 16384;                      // Ph -> Pph -> Kh
    _Float16* H2 = H1 + 16384;                      // T1h -> T2h -> T3h
    float* mv    = smem + 3 * 16384 + 3 * 8192;     // after 3x32KB f16
    float* mp    = mv + D;
    float* innov = mp + D;
    float* alph  = innov + D;
    float* pers  = alph + D;
    float* scal  = pers + D;   // [0]=chol jitter, [1]=logdet, [2]=mix

    const int tid = threadIdx.x;
    const int b   = blockIdx.x;

    if (tid < D) {
        float s = 1.0f / (1.0f + expf(-praw[tid]));
        pers[tid] = 0.6f + (0.98f - 0.6f) * s;
        mv[tid] = san(im[tid]);
    }
    if (tid == 0) {
        float s = 1.0f / (1.0f + expf(-graw[0]));
        scal[2] = 0.35f + (1.0f - 0.35f) * s;
    }
    for (int idx = tid; idx < D * D; idx += 256) {
        int i = idx >> 7, j = idx & (D - 1);
        Pm[idx] = (i == j) ? (expf(ild[i]) + 1.0e-6f) : 0.0f;
    }
    __syncthreads();
    const float mix = scal[2], omix = 1.0f - mix;

    for (int t = 0; t < TSTEPS; ++t) {
        const float* Fr = Fraw + (size_t)(b * TSTEPS + t) * (D * D);
        const float* y  = obs  + (size_t)(b * TSTEPS + t) * D;

        // Fh -> H0, Ph (P symmetric) -> H1
        for (int idx = tid; idx < D * D; idx += 256) {
            int i = idx >> 7, j = idx & (D - 1);
            float v = mix * san(Fr[idx]);
            if (i == j) v += omix * pers[i];
            H0[idx] = to_h(san(v));
            H1[idx] = to_h(Pm[idx]);
        }
        if (t + 1 < TSTEPS)   // hide next step's 64KB F stream behind compute
            __builtin_prefetch(Fr + D * D + tid * 64, 0, 1);
        __syncthreads();

        // m_pred = F m ; innov = san(y - m_pred)
        if (tid < D) {
            float s = 0.0f;
            for (int k = 0; k < D; ++k) s += (float)H0[tid * D + k] * mv[k];
            mp[tid] = s;
            innov[tid] = san(y[tid] - s);
        }
        __syncthreads();

        mm128h(H0, H1, H2, nullptr, nullptr, false, false);   // T1 = F P        -> H2
        __syncthreads();
        // S0 = san(T1 F^T + Q) -> Lb (f32) ; P_pred mirror -> H1
        mm128h(H2, H0, H1, Lb, Qg, false, true);
        __syncthreads();
        // symmetrize S0, add R, sanitize
        if (tid < D) {
            for (int j = tid; j < D; ++j) {
                float v = 0.5f * (Lb[tid * D + j] + Lb[j * D + tid]);
                Lb[tid * D + j] = v; Lb[j * D + tid] = v;
            }
        }
        __syncthreads();
        for (int idx = tid; idx < D * D; idx += 256)
            Lb[idx] = san(Lb[idx] + Rg[idx]);
        __syncthreads();
        if (tid == 0) {
            float s = 0.0f;
            for (int i = 0; i < D; ++i) s += fabsf(Lb[i * D + i]);
            scal[0] = 1.0e-6f * fmaxf(s * (1.0f / D), 1.0f);
        }
        __syncthreads();
        if (tid < D) Lb[tid * D + tid] += scal[0];
        __syncthreads();

        // Cholesky in place (lower) on Lb; per-thread local sqrt/rcp
        for (int j = 0; j < D; ++j) {
            const float ajj = Lb[j * D + j];
            __syncthreads();                       // reads before writes
            const float ljj = sqrtf(fmaxf(ajj, 1.0e-30f));
            const float inv = 1.0f / ljj;
            for (int i = j + tid; i < D; i += 256)
                Lb[i * D + j] = (i == j) ? ljj : Lb[i * D + j] * inv;
            __syncthreads();
            for (int c = j + 1; c < D; ++c) {
                const float ljc = Lb[c * D + j];
                for (int r = c + tid; r < D; r += 256)
                    Lb[r * D + c] -= Lb[r * D + j] * ljc;
            }
            __syncthreads();
        }

        if (tid == 0) {
            float s = 0.0f;
            for (int j = 0; j < D; ++j) s += logf(Lb[j * D + j]);
            scal[1] = 2.0f * s;
        }
        // Solves: thread c<128 -> row c of K = (S^-1 P_pred)^T into Kf
        // (RHS = P_pred column c = row c of symmetric mirror H1); thread 128 -> alpha
        if (tid <= D) {
            float* x = (tid < D) ? (Kf + tid * D) : alph;
            for (int j = 0; j < D; ++j) {                 // L z = rhs
                float s = (tid < D) ? (float)H1[tid * D + j] : innov[j];
                for (int k = 0; k < j; ++k) s -= Lb[j * D + k] * x[k];
                x[j] = s / Lb[j * D + j];
            }
            for (int j = D - 1; j >= 0; --j) {            // L^T w = z
                float s = x[j];
                for (int k = j + 1; k < D; ++k) s -= Lb[k * D + j] * x[k];
                x[j] = s / Lb[j * D + j];
            }
        }
        __syncthreads();

        if (tid == 0) {   // per-step NLL
            float dot = 0.0f;
            for (int i = 0; i < D; ++i) dot += innov[i] * alph[i];
            float nll = 0.5f * (dot + scal[1] + D * 1.8378770664093453f) / D;
            out[b * TSTEPS + t] = fminf(nll, 10000.0f);
        }
        if (tid < D) {    // m_new = san(m_pred + K innov)  (f32 K)
            float s = 0.0f;
            for (int k = 0; k < D; ++k) s += Kf[tid * D + k] * innov[k];
            mv[tid] = san(mp[tid] + s);
        }
        for (int idx = tid; idx < D * D; idx += 256) {    // IKh = I - K -> H0
            int i = idx >> 7, j = idx & (D - 1);
            H0[idx] = to_h(((i == j) ? 1.0f : 0.0f) - Kf[idx]);
        }
        __syncthreads();

        // Joseph form: P = IK P_pred IK^T + K R K^T
        mm128h(H0, H1, H2, nullptr, nullptr, false, false);   // T2 = IK P_pred -> H2
        __syncthreads();
        mm128h(H2, H0, nullptr, Pm, nullptr, false, false);   // P  = T2 IK^T   -> Pm
        __syncthreads();
        for (int idx = tid; idx < D * D; idx += 256)          // Kh -> H1
            H1[idx] = to_h(Kf[idx]);
        __syncthreads();
        mm128h(H1, Rh, H2, nullptr, nullptr, false, false);   // T3 = K R (R sym, L2) -> H2
        __syncthreads();
        mm128h(H2, H1, nullptr, Pm, nullptr, true, false);    // P += T3 K^T
        __syncthreads();
        // P = sym(san(P))
        if (tid < D) {
            for (int j = tid; j < D; ++j) {
                float v = 0.5f * (san(Pm[tid * D + j]) + san(Pm[j * D + tid]));
                Pm[tid * D + j] = v; Pm[j * D + tid] = v;
            }
        }
        __syncthreads();
    }
}

extern "C" void kernel_launch(void* const* d_in, const int* in_sizes, int n_in,
                              void* d_out, int out_size, void* d_ws, size_t ws_size,
                              hipStream_t stream)
{
    const float* obs  = (const float*)d_in[0];  // [8,256,128]
    const float* Fraw = (const float*)d_in[1];  // [8,256,128,128]
    const float* qf   = (const float*)d_in[2];  // [128,128]
    const float* rf   = (const float*)d_in[3];  // [128,128]
    const float* im   = (const float*)d_in[4];  // [128]
    const float* ild  = (const float*)d_in[5];  // [128]
    const float* praw = (const float*)d_in[6];  // [128]
    const float* graw = (const float*)d_in[7];  // [1]
    float* out = (float*)d_out;                 // [8,256]

    float* Qg = (float*)d_ws;
    float* Rg = Qg + D * D;
    _Float16* Rh = (_Float16*)(Rg + D * D);

    kf_build_qr<<<dim3(128), dim3(256), 0, stream>>>(qf, rf, Qg, Rg, Rh);

    // 3 x 64KB f32 + 3 x 32KB f16 + scalars  (~297 KB < 320 KB WGP LDS)
    const size_t smem = (size_t)(3 * D * D + 3 * (D * D / 2) + 5 * D + 8) * sizeof(float);
    kf_filter<<<dim3(NB), dim3(256), smem, stream>>>(obs, Fraw, im, ild, praw, graw,
                                                     Qg, Rg, Rh, out);
}
